// Model_76527727280253
// MI455X (gfx1250) — matrix operand order repaired and checked by reading, not compiled, not run on hardware
//
#include <hip/hip_runtime.h>

typedef _Float16 h16 __attribute__((ext_vector_type(16)));
typedef _Float16 h8  __attribute__((ext_vector_type(8)));
typedef float    f8  __attribute__((ext_vector_type(8)));
typedef unsigned int u32x4 __attribute__((ext_vector_type(4)));
typedef int          i32x4 __attribute__((ext_vector_type(4)));
typedef int          i32x8 __attribute__((ext_vector_type(8)));

#define DIMS 64

// Tensor Data Mover availability (device pass only; host pass falls back)
#if defined(__AMDGCN__) && defined(__has_builtin)
#  if __has_builtin(__builtin_amdgcn_tensor_load_to_lds) && __has_builtin(__builtin_amdgcn_s_wait_tensorcnt)
#    define HAVE_TDM 1
#  endif
#endif
#ifndef HAVE_TDM
#  define HAVE_TDM 0
#endif

// ---------------- utility kernels ----------------

__global__ void zero_f32_kernel(float* __restrict__ p, long n) {
    long i = (long)blockIdx.x * blockDim.x + threadIdx.x;
    if (i < n) p[i] = 0.f;
}

__global__ void zero_i32_kernel(int* __restrict__ p, int n) {
    int i = blockIdx.x * blockDim.x + threadIdx.x;
    if (i < n) p[i] = 0;
}

__global__ void deg_kernel(const int* __restrict__ dst, float* __restrict__ deg, int E) {
    int e = blockIdx.x * blockDim.x + threadIdx.x;
    if (e < E) atomicAdd(&deg[dst[e]], 1.0f);
}

// ---------------- counting sort of edges by etype ----------------

__global__ void hist_kernel(const int* __restrict__ et, int* __restrict__ cnt, int E) {
    int e = blockIdx.x * blockDim.x + threadIdx.x;
    if (e < E) atomicAdd(&cnt[et[e]], 1);
}

__global__ void scan_kernel(const int* __restrict__ cnt, int* __restrict__ starts,
                            int* __restrict__ cursor, int R) {
    if (blockIdx.x == 0 && threadIdx.x == 0) {
        int acc = 0;
        for (int r = 0; r < R; ++r) { starts[r] = acc; cursor[r] = acc; acc += cnt[r]; }
        starts[R] = acc;
    }
}

__global__ void permute_kernel(const int* __restrict__ et, int* __restrict__ cursor,
                               int* __restrict__ eperm, int E) {
    int e = blockIdx.x * blockDim.x + threadIdx.x;
    if (e < E) { int pos = atomicAdd(&cursor[et[e]], 1); eperm[pos] = e; }
}

// ---------------- layer 0 (in_dim == 1) ----------------

// T0[n, r] = sum over edges (etype==r, dst==n) of deg[src]
__global__ void t0_kernel(const int* __restrict__ src, const int* __restrict__ dst,
                          const int* __restrict__ et, const float* __restrict__ deg,
                          float* __restrict__ T0, int E, int R) {
    int e = blockIdx.x * blockDim.x + threadIdx.x;
    if (e < E) atomicAdd(&T0[(size_t)dst[e] * R + et[e]], deg[src[e]]);
}

// h[n,o] = relu( sum_r T0[n,r]*W0[r,0,o] + deg[n]*L0[0,o] + b0[o] )
__global__ void layer0_kernel(const float* __restrict__ T0, const float* __restrict__ deg,
                              const float* __restrict__ W0, const float* __restrict__ L0,
                              const float* __restrict__ b0, float* __restrict__ h,
                              int N, int R) {
    int i = blockIdx.x * blockDim.x + threadIdx.x;
    if (i >= N * DIMS) return;
    int n = i >> 6, o = i & 63;
    float s = deg[n] * L0[o] + b0[o];
    for (int r = 0; r < R; ++r) s += T0[(size_t)n * R + r] * W0[r * DIMS + o];
    h[i] = fmaxf(s, 0.f);
}

// ---------------- precision conversions ----------------

__global__ void convh_kernel(const float* __restrict__ h, _Float16* __restrict__ Hh,
                             long nValid, long nTot) {
    long i = (long)blockIdx.x * blockDim.x + threadIdx.x;
    if (i >= nTot) return;
    Hh[i] = (i < nValid) ? (_Float16)h[i] : (_Float16)0.f;
}

// W [nmat,64,64] f32 (d-major) -> WT [nmat,64,64] f16 (o-major): WT[m][o*64+d] = W[m][d*64+o]
__global__ void convT_kernel(const float* __restrict__ W, _Float16* __restrict__ WT, int nmat) {
    int i = blockIdx.x * blockDim.x + threadIdx.x;
    if (i >= nmat * 4096) return;
    int m = i >> 12, rem = i & 4095, d = rem >> 6, o = rem & 63;
    WT[(m << 12) + (o << 6) + d] = (_Float16)W[i];
}

// ---------------- WMMA GEMM: P[n,o] = sum_d Hh[n,d] * WT[o*64+d] ----------------
// The 8KB weight tile is staged into LDS once per workgroup — via the Tensor Data
// Mover (tensor_load_to_lds + s_wait_tensorcnt) when available, else cooperative
// b128 copies. Each wave then computes a 16x64 tile with 8x v_wmma_f32_16x16x32_f16,
// pulling B fragments from LDS (ds_load_b128) and A fragments from global (b128).
__global__ void __launch_bounds__(256)
gemm16_kernel(const _Float16* __restrict__ Hh, const _Float16* __restrict__ WT,
              float* __restrict__ Pout, int nTiles) {
    __shared__ _Float16 Wlds[DIMS * DIMS];   // 8 KB

    int wid  = threadIdx.x >> 5;
    int lane = threadIdx.x & 31;

#if HAVE_TDM
    if (wid == 0) {
        // Tensor DMA descriptor: contiguous 4096 x 2B tile (one row).
        unsigned long long ga = (unsigned long long)(uintptr_t)WT;
        unsigned ldsOff = (unsigned)(uintptr_t)(&Wlds[0]);
        u32x4 g0;
        g0[0] = 1u;                                             // count=1, user mode
        g0[1] = ldsOff;                                         // lds_addr
        g0[2] = (unsigned)(ga & 0xFFFFFFFFu);                   // global_addr[31:0]
        g0[3] = (unsigned)((ga >> 32) & 0x01FFFFFFu) | 0x80000000u; // addr[56:32] | type=2
        const unsigned TD0 = 4096u, TD1 = 1u, TILE0 = 4096u, TILE1 = 1u;
        const unsigned long long S0 = 4096ull;
        i32x8 g1;
        g1[0] = (int)(1u << 16);                                // data_size=1 (2 bytes)
        g1[1] = (int)((TD0 & 0xFFFFu) << 16);                   // tensor_dim0 lo
        g1[2] = (int)(((TD0 >> 16) & 0xFFFFu) | ((TD1 & 0xFFFFu) << 16));
        g1[3] = (int)(((TD1 >> 16) & 0xFFFFu) | ((TILE0 & 0xFFFFu) << 16)); // tile_dim0
        g1[4] = (int)(TILE1 & 0xFFFFu);                         // tile_dim1; tile_dim2=0
        g1[5] = (int)(S0 & 0xFFFFFFFFull);                      // dim0 stride lo
        g1[6] = (int)((S0 >> 32) & 0xFFFFull);                  // stride hi | dim1 stride lo
        g1[7] = 0;
        i32x4 z4 = {0, 0, 0, 0};
#if defined(__clang_major__) && (__clang_major__ >= 23)
        i32x8 z8 = {0, 0, 0, 0, 0, 0, 0, 0};
        __builtin_amdgcn_tensor_load_to_lds(g0, g1, z4, z4, z8, 0);
#else
        __builtin_amdgcn_tensor_load_to_lds(g0, g1, z4, z4, 0);
#endif
        __builtin_amdgcn_s_wait_tensorcnt(0);
    }
    __syncthreads();
#else
    // fallback: cooperative global->LDS copy, 16B per thread per step
    for (int c = threadIdx.x; c < (DIMS * DIMS * 2) / 16; c += 256)
        *reinterpret_cast<uint4*>((char*)Wlds + (size_t)c * 16) =
            *reinterpret_cast<const uint4*>((const char*)WT + (size_t)c * 16);
    __syncthreads();
#endif

    int wave = blockIdx.x * 8 + wid;
    if (wave < nTiles) {                     // wave-uniform: EXEC stays all-ones
        int m  = lane & 15;
        int hi = lane >> 4;

        const _Float16* Arow = Hh + (size_t)(wave * 16 + m) * DIMS;
        f8 acc[4] = {};

#pragma unroll
        for (int kb = 0; kb < 2; ++kb) {
            int ka = kb * 32 + hi * 8;
            h8 a0 = *reinterpret_cast<const h8*>(Arow + ka);
            h8 a1 = *reinterpret_cast<const h8*>(Arow + ka + 16);
            h16 a;
#pragma unroll
            for (int i = 0; i < 8; ++i) { a[i] = a0[i]; a[i + 8] = a1[i]; }
#pragma unroll
            for (int j = 0; j < 4; ++j) {
                const _Float16* Bp = &Wlds[(j * 16 + m) * DIMS + kb * 32 + hi * 8];
                h8 b0v = *reinterpret_cast<const h8*>(Bp);
                h8 b1v = *reinterpret_cast<const h8*>(Bp + 16);
                h16 b;
#pragma unroll
                for (int i = 0; i < 8; ++i) { b[i] = b0v[i]; b[i + 8] = b1v[i]; }
                acc[j] = __builtin_amdgcn_wmma_f32_16x16x32_f16(
                    false, a, false, b, (short)0, acc[j], false, false);
            }
        }

        // D layout: lanes 0-15 -> N=lane, M=i ; lanes 16-31 -> N=lane-16, M=i+8
        float* Prow = Pout + (size_t)wave * 16 * DIMS;
#pragma unroll
        for (int j = 0; j < 4; ++j)
#pragma unroll
            for (int i = 0; i < 8; ++i)
                Prow[(size_t)(hi * 8 + i) * DIMS + j * 16 + m] = acc[j][i];
    }
}

// ---------------- sorted edge scatter: agg[dst,:] += P[src,:] for etype==r ----------------
// Persistent grid; relation's edge range is read from device memory (graph-capture safe).
__global__ void scatter_sorted_kernel(const int* __restrict__ starts, const int* __restrict__ eperm,
                                      const int* __restrict__ src, const int* __restrict__ dst,
                                      const float* __restrict__ P, float* __restrict__ agg, int r) {
    int s0 = starts[r], s1 = starts[r + 1];
    long total  = (long)(s1 - s0) << 6;
    long stride = (long)gridDim.x * blockDim.x;
    for (long idx = (long)blockIdx.x * blockDim.x + threadIdx.x; idx < total; idx += stride) {
        int e = eperm[s0 + (int)(idx >> 6)];
        int d = (int)(idx & 63);
        atomicAdd(&agg[(size_t)dst[e] * DIMS + d], P[(size_t)src[e] * DIMS + d]);
    }
}

__global__ void bias_act_kernel(const float* __restrict__ agg, const float* __restrict__ bias,
                                float* __restrict__ hout, long n, int act) {
    long i = (long)blockIdx.x * blockDim.x + threadIdx.x;
    if (i >= n) return;
    float v = agg[i] + bias[i & 63];
    hout[i] = act ? fmaxf(v, 0.f) : v;
}

// ---------------- pooling head ----------------

// emb[n,:] = concat(h[n,:], nuc[n]);  gate[n] = emb[n,:] . gate_w + gate_b
__global__ void emb_gate_kernel(const float* __restrict__ h, const float* __restrict__ nuc,
                                const float* __restrict__ gw, const float* __restrict__ gb,
                                float* __restrict__ emb, float* __restrict__ gate,
                                int N, int EMB) {
    int n = blockIdx.x * blockDim.x + threadIdx.x;
    if (n >= N) return;
    float g = gb[0];
    float* er = emb + (size_t)n * EMB;
    const float* hr = h + (size_t)n * DIMS;
    for (int j = 0; j < EMB - 1; ++j) { float v = hr[j]; er[j] = v; g += v * gw[j]; }
    float vn = nuc[n]; er[EMB - 1] = vn; g += vn * gw[EMB - 1];
    gate[n] = g;
}

// one block per graph: segment softmax over sorted node2graph + weighted sum of emb
__global__ void __launch_bounds__(128)
pool_kernel(const int* __restrict__ n2g, const float* __restrict__ gate,
            const float* __restrict__ emb, float* __restrict__ fp, int N, int EMB) {
    __shared__ float red[128];
    __shared__ int s_lo, s_hi;
    __shared__ float s_max, s_den;
    int g = blockIdx.x, tid = threadIdx.x;
    if (tid == 0) {
        int lo = 0, hb = N;
        while (lo < hb) { int mid = (lo + hb) >> 1; if (n2g[mid] < g) lo = mid + 1; else hb = mid; }
        s_lo = lo;
        int lo2 = lo, hb2 = N;
        while (lo2 < hb2) { int mid = (lo2 + hb2) >> 1; if (n2g[mid] < g + 1) lo2 = mid + 1; else hb2 = mid; }
        s_hi = lo2;
    }
    __syncthreads();
    int lo = s_lo, hi = s_hi;

    float m = -3.4e38f;
    for (int n = lo + tid; n < hi; n += 128) m = fmaxf(m, gate[n]);
    red[tid] = m; __syncthreads();
    for (int s = 64; s > 0; s >>= 1) { if (tid < s) red[tid] = fmaxf(red[tid], red[tid + s]); __syncthreads(); }
    if (tid == 0) s_max = red[0];
    __syncthreads();
    float gm = s_max;

    float sum = 0.f;
    for (int n = lo + tid; n < hi; n += 128) sum += expf(gate[n] - gm);
    red[tid] = sum; __syncthreads();
    for (int s = 64; s > 0; s >>= 1) { if (tid < s) red[tid] += red[tid + s]; __syncthreads(); }
    if (tid == 0) s_den = red[0];
    __syncthreads();
    float den = s_den;

    for (int j = tid; j < EMB; j += 128) {
        float a = 0.f;
        for (int n = lo; n < hi; ++n) a += expf(gate[n] - gm) * emb[(size_t)n * EMB + j];
        fp[(size_t)g * EMB + j] = (hi > lo) ? (a / den) : 0.f;
    }
}

__global__ void mlp1_kernel(const float* __restrict__ fp, const float* __restrict__ A1,
                            const float* __restrict__ ab1, float* __restrict__ h1,
                            int G, int EMB, int HID) {
    int i = blockIdx.x * blockDim.x + threadIdx.x;
    if (i >= G * HID) return;
    int g = i / HID, j = i % HID;
    float s = ab1[j];
    for (int d = 0; d < EMB; ++d) s += fp[(size_t)g * EMB + d] * A1[(size_t)d * HID + j];
    h1[i] = fmaxf(s, 0.f);
}

__global__ void mlp2_kernel(const float* __restrict__ h1, const float* __restrict__ A2,
                            const float* __restrict__ ab2, float* __restrict__ out,
                            int G, int HID, int FP) {
    int i = blockIdx.x * blockDim.x + threadIdx.x;
    if (i >= G * FP) return;
    int g = i / FP, k = i % FP;
    float s = ab2[k];
    for (int d = 0; d < HID; ++d) s += h1[(size_t)g * HID + d] * A2[(size_t)d * FP + k];
    out[i] = 1.f / (1.f + expf(-s));
}

// ---------------- host launcher ----------------

extern "C" void kernel_launch(void* const* d_in, const int* in_sizes, int n_in,
                              void* d_out, int out_size, void* d_ws, size_t ws_size,
                              hipStream_t stream) {
    const int*   src = (const int*)d_in[0];
    const int*   dst = (const int*)d_in[1];
    const int*   et  = (const int*)d_in[2];
    const int*   n2g = (const int*)d_in[3];
    const float* nuc = (const float*)d_in[4];
    const float* W0  = (const float*)d_in[5];
    const float* b0  = (const float*)d_in[6];
    const float* L0  = (const float*)d_in[7];
    const float* W1  = (const float*)d_in[8];
    const float* b1  = (const float*)d_in[9];
    const float* L1  = (const float*)d_in[10];
    const float* W2  = (const float*)d_in[11];
    const float* b2f = (const float*)d_in[12];
    const float* L2  = (const float*)d_in[13];
    const float* gw  = (const float*)d_in[14];
    const float* gb  = (const float*)d_in[15];
    const float* A1  = (const float*)d_in[16];
    const float* ab1 = (const float*)d_in[17];
    const float* A2  = (const float*)d_in[18];
    const float* ab2 = (const float*)d_in[19];
    (void)n_in; (void)ws_size;

    const int E   = in_sizes[0];
    const int N   = in_sizes[4];
    const int R   = in_sizes[5] / DIMS;     // W0 is [R,1,64]
    const int EMB = in_sizes[14];           // 65
    const int HID = in_sizes[17];           // 128
    const int FP  = in_sizes[19];           // 166
    const int G   = (out_size - N * EMB) / FP;

    const int nTiles = (N + 15) / 16;
    const int nPad   = nTiles * 16;

    // workspace carve-up (256B aligned)
    size_t off = 0;
    auto alloc = [&](size_t bytes) -> char* {
        char* p = (char*)d_ws + off;
        off = (off + bytes + 255) & ~(size_t)255;
        return p;
    };
    float*    deg    = (float*)alloc((size_t)N * 4);
    float*    T0     = (float*)alloc((size_t)N * R * 4);
    float*    hA     = (float*)alloc((size_t)nPad * DIMS * 4);
    float*    hB     = (float*)alloc((size_t)nPad * DIMS * 4);
    float*    agg    = (float*)alloc((size_t)nPad * DIMS * 4);
    float*    P      = (float*)alloc((size_t)nPad * DIMS * 4);
    _Float16* Hh     = (_Float16*)alloc((size_t)nPad * DIMS * 2);
    _Float16* W1hT   = (_Float16*)alloc((size_t)R * DIMS * DIMS * 2);
    _Float16* W2hT   = (_Float16*)alloc((size_t)R * DIMS * DIMS * 2);
    _Float16* L1hT   = (_Float16*)alloc((size_t)DIMS * DIMS * 2);
    _Float16* L2hT   = (_Float16*)alloc((size_t)DIMS * DIMS * 2);
    float*    gate   = (float*)alloc((size_t)N * 4);
    float*    fp     = (float*)alloc((size_t)G * EMB * 4);
    float*    h1     = (float*)alloc((size_t)G * HID * 4);
    int*      cnt    = (int*)alloc((size_t)R * 4);
    int*      starts = (int*)alloc((size_t)(R + 1) * 4);
    int*      cursor = (int*)alloc((size_t)R * 4);
    int*      eperm  = (int*)alloc((size_t)E * 4);

    float* out = (float*)d_out;             // [G, FP]
    float* emb = out + (size_t)G * FP;      // [N, EMB]

    const unsigned gE    = (unsigned)((E + 255) / 256);
    const unsigned gN64  = (unsigned)(((long)N * DIMS + 255) / 256);
    const unsigned gPad  = (unsigned)(((long)nPad * DIMS + 255) / 256);
    const unsigned gGem  = (unsigned)((nTiles + 7) / 8);
    const unsigned gScat = 2048;            // persistent grid for scatter

    // weight conversions (f32 -> f16, transposed to o-major for B fragments)
    convT_kernel<<<(unsigned)((R * 4096 + 255) / 256), 256, 0, stream>>>(W1, W1hT, R);
    convT_kernel<<<(unsigned)((R * 4096 + 255) / 256), 256, 0, stream>>>(W2, W2hT, R);
    convT_kernel<<<(unsigned)((4096 + 255) / 256), 256, 0, stream>>>(L1, L1hT, 1);
    convT_kernel<<<(unsigned)((4096 + 255) / 256), 256, 0, stream>>>(L2, L2hT, 1);

    // counting sort of edges by etype (done once; reused by both conv layers)
    zero_i32_kernel<<<(unsigned)((R + 63) / 64), 64, 0, stream>>>(cnt, R);
    hist_kernel<<<gE, 256, 0, stream>>>(et, cnt, E);
    scan_kernel<<<1, 32, 0, stream>>>(cnt, starts, cursor, R);
    permute_kernel<<<gE, 256, 0, stream>>>(et, cursor, eperm, E);

    // in-degree feature
    zero_f32_kernel<<<(unsigned)((N + 255) / 256), 256, 0, stream>>>(deg, (long)N);
    deg_kernel<<<gE, 256, 0, stream>>>(dst, deg, E);

    // layer 0 (in_dim = 1): pre-aggregate per (dst, rel), then contract with W0
    zero_f32_kernel<<<(unsigned)(((long)N * R + 255) / 256), 256, 0, stream>>>(T0, (long)N * R);
    t0_kernel<<<gE, 256, 0, stream>>>(src, dst, et, deg, T0, E, R);
    layer0_kernel<<<gN64, 256, 0, stream>>>(T0, deg, W0, L0, b0, hA, N, R);

    // layers 1 & 2: per-relation WMMA GEMM + sorted edge scatter
    auto run_layer = [&](const float* hin, const _Float16* WhT, const _Float16* LhT,
                         const float* bias, float* hout, int act) {
        convh_kernel<<<gPad, 256, 0, stream>>>(hin, Hh, (long)N * DIMS, (long)nPad * DIMS);
        // agg = h @ L (self-loop), written directly — no zero pass needed
        gemm16_kernel<<<gGem, 256, 0, stream>>>(Hh, LhT, agg, nTiles);
        for (int r = 0; r < R; ++r) {
            gemm16_kernel<<<gGem, 256, 0, stream>>>(Hh, WhT + (size_t)r * 4096, P, nTiles);
            scatter_sorted_kernel<<<gScat, 256, 0, stream>>>(starts, eperm, src, dst, P, agg, r);
        }
        bias_act_kernel<<<gN64, 256, 0, stream>>>(agg, bias, hout, (long)N * DIMS, act);
    };
    run_layer(hA, W1hT, L1hT, b1, hB, 1);
    run_layer(hB, W2hT, L2hT, b2f, hA, 0);

    // embedding + gate, pooling, attributor MLP
    emb_gate_kernel<<<(unsigned)((N + 255) / 256), 256, 0, stream>>>(hA, nuc, gw, gb, emb, gate, N, EMB);
    pool_kernel<<<(unsigned)G, 128, 0, stream>>>(n2g, gate, emb, fp, N, EMB);
    mlp1_kernel<<<(unsigned)((G * HID + 255) / 256), 256, 0, stream>>>(fp, A1, ab1, h1, G, EMB, HID);
    mlp2_kernel<<<(unsigned)((G * FP + 255) / 256), 256, 0, stream>>>(h1, A2, ab2, out, G, HID, FP);
}
